// NewNet_7584912245326
// MI455X (gfx1250) — matrix operand order repaired and checked by reading, not compiled, not run on hardware
//
#include <hip/hip_runtime.h>
#include <hip/hip_bf16.h>
#include <math.h>

typedef __attribute__((ext_vector_type(16))) _Float16 v16h;
typedef __attribute__((ext_vector_type(8)))  _Float16 v8h;
typedef __attribute__((ext_vector_type(4)))  _Float16 v4h;
typedef __attribute__((ext_vector_type(8)))  float    v8f;

#define HEADS  3
#define HDIM   32
#define DIM    96
#define QKVN   288
#define IMG    224
#define NW     32
#define TOK    49
#define SCALEF 0.17677669529663687f

// ---- dynamic LDS layout (bytes); all fragment bases 16B-aligned ----
#define PITCH_X   104                      // halves (208B = 16*13)
#define OFF_X     0
#define SZ_X      (64*PITCH_X*2)           // 13312
#define OFF_W     (OFF_X + SZ_X)           // 13312
#define SZ_W      (QKVN*DIM*2)             // 55296 (w_qkv^T f16; reused for w_out^T)
#define OFF_QKV   (OFF_W + SZ_W)           // 68608  (q,k only: cols 0..191)
#define PITCH_QKV 200                      // 400B = 16*25
#define SZ_QKV    (64*PITCH_QKV*2)         // 25600
#define OFF_VT    (OFF_QKV + SZ_QKV)       // 94208  (V^T: [96][64] tokens)
#define PITCH_VT  72                       // 144B = 16*9
#define SZ_VT     (DIM*PITCH_VT*2)         // 13824
#define OFF_DOTS  (OFF_VT + SZ_VT)         // 108032
#define PITCH_D   65
#define SZ_DOTS   (64*PITCH_D*4)           // 16640
#define OFF_ATTN  (OFF_DOTS + SZ_DOTS)     // 124672
#define PITCH_A   72
#define SZ_ATTN   (64*PITCH_A*2)           // 9216
#define OFF_CTX   (OFF_ATTN + SZ_ATTN)     // 133888
#define PITCH_C   104
#define SZ_CTX    (64*PITCH_C*2)           // 13312
#define OFF_BIAS  (OFF_CTX + SZ_CTX)       // 147200 (pos-bias + masks, f32 [49][52])
#define PITCH_B   52
#define SZ_BIAS   (TOK*PITCH_B*4 + 48)     // 10240
#define OFF_INV   (OFF_BIAS + SZ_BIAS)     // 157440 (1/rowsum, 64 f32)
#define SZ_INV    256
#define OFF_POS   (OFF_INV + SZ_INV)       // 157696
#define SZ_POS    (176*4)                  // 704
#define SMEM_TOTAL (OFF_POS + SZ_POS)      // 158400

__device__ inline v8f wmma_f16(v16h a, v16h b, v8f c) {
  return __builtin_amdgcn_wmma_f32_16x16x32_f16(false, a, false, b, (short)0, c,
                                                false, false);
}

__device__ inline v16h cat16(v8h lo, v8h hi) {
  return __builtin_shufflevector(lo, hi, 0, 1, 2, 3, 4, 5, 6, 7,
                                 8, 9, 10, 11, 12, 13, 14, 15);
}

// A fragment (16x32 f16): lane row m = m0+lane%16; halves 0..7 = [k0+ko..],
// 8..15 = [k0+16+ko..] ; two 16B LDS loads
__device__ inline v16h load_A(const _Float16* base, int pitch, int m0, int k0, int lane) {
  int m = m0 + (lane & 15);
  int ko = (lane >> 4) * 8;
  const _Float16* p = base + m * pitch + k0 + ko;
  return cat16(*(const v8h*)p, *(const v8h*)(p + 16));
}

// B fragment (32x16 f16) from a K-contiguous (transposed / row-of-K) source:
// lane col n = n0+lane%16 selects row; halves 0..7 = [k0+kb..], 8..15 = [k0+kb+16..]
__device__ inline v16h load_Brow(const _Float16* base, int pitch, int n0, int k0, int lane) {
  int n = n0 + (lane & 15);
  int kb = k0 + ((lane >> 4) << 3);
  const _Float16* p = base + n * pitch + kb;
  return cat16(*(const v8h*)p, *(const v8h*)(p + 16));
}

__device__ inline void store_D_h(_Float16* base, int pitch, int m0, int n0, int lane,
                                 v8f acc, float scale) {
  int n = n0 + (lane & 15);
  int m = m0 + ((lane >> 4) << 3);
#pragma unroll
  for (int i = 0; i < 8; ++i) base[(m + i) * pitch + n] = (_Float16)(acc[i] * scale);
}

// transposed D store: dest[n][m0..m0+7] -> 8 contiguous halves, one 16B store
__device__ inline void store_D_hT(_Float16* base, int pitch, int m0, int n0, int lane, v8f acc) {
  int n = n0 + (lane & 15);
  int m = m0 + ((lane >> 4) << 3);
  v8h hv;
#pragma unroll
  for (int i = 0; i < 8; ++i) hv[i] = (_Float16)acc[i];
  *(v8h*)(base + n * pitch + m) = hv;
}

__device__ inline void store_D_f(float* base, int pitch, int m0, int n0, int lane, v8f acc) {
  int n = n0 + (lane & 15);
  int m = m0 + ((lane >> 4) << 3);
#pragma unroll
  for (int i = 0; i < 8; ++i) base[(m + i) * pitch + n] = acc[i];
}

// ------------- prep: transpose weights to f16 N-major, zero accumulators -------------
__global__ void prep_kernel(const float* __restrict__ wqkv, const float* __restrict__ wout,
                            _Float16* __restrict__ wqkvT, _Float16* __restrict__ woutT,
                            float* __restrict__ sums, unsigned* __restrict__ maxe) {
  int t = blockIdx.x * blockDim.x + threadIdx.x;
  if (t < QKVN * DIM) {                 // wqkvT[n][k] = wqkv[k][n]
    int n = t / DIM, k = t - n * DIM;
    wqkvT[t] = (_Float16)wqkv[k * QKVN + n];
  }
  if (t < DIM * DIM) {                  // woutT[n][k] = wout[k][n]
    int n = t / DIM, k = t - n * DIM;
    woutT[t] = (_Float16)wout[k * DIM + n];
  }
  if (t < 768) { sums[t] = 0.f; maxe[t] = 0u; }
}

// ------------- fused shifted-window attention + out-proj -------------
__global__ void __launch_bounds__(128)
swin_attn_kernel(const float* __restrict__ x, const float* __restrict__ pos_emb,
                 const float* __restrict__ b_out, const _Float16* __restrict__ wqkvT,
                 const _Float16* __restrict__ woutT, float* __restrict__ out) {
  extern __shared__ char smem[];
  _Float16* sX    = (_Float16*)(smem + OFF_X);
  _Float16* sW    = (_Float16*)(smem + OFF_W);
  _Float16* sQKV  = (_Float16*)(smem + OFF_QKV);
  _Float16* sVt   = (_Float16*)(smem + OFF_VT);
  float*    sDots = (float*)   (smem + OFF_DOTS);
  _Float16* sAttn = (_Float16*)(smem + OFF_ATTN);
  _Float16* sCtx  = (_Float16*)(smem + OFF_CTX);
  float*    sBias = (float*)   (smem + OFF_BIAS);
  float*    sInv  = (float*)   (smem + OFF_INV);
  float*    sPos  = (float*)   (smem + OFF_POS);

  const int tid  = threadIdx.x;
  const int wave = tid >> 5;
  const int lane = tid & 31;

  const int bb = blockIdx.x >> 10;
  const int wf = blockIdx.x & 1023;
  const int wy = wf >> 5, wx = wf & 31;
  const bool maskUL = (wy == NW - 1);
  const bool maskLR = (wx == NW - 1);

  // ---- stage shifted x window (float4 -> 4 halves) ----
  for (int t = tid; t < TOK * 24; t += 128) {
    int r = t / 24, c4 = (t - r * 24) * 4;
    int rr = r / 7, cc = r - rr * 7;
    int gh = wy * 7 + rr + 3; if (gh >= IMG) gh -= IMG;
    int gw = wx * 7 + cc + 3; if (gw >= IMG) gw -= IMG;
    const float4 v = *(const float4*)(x + (((size_t)bb * IMG + gh) * IMG + gw) * DIM + c4);
    v4h hv; hv[0] = (_Float16)v.x; hv[1] = (_Float16)v.y;
    hv[2] = (_Float16)v.z; hv[3] = (_Float16)v.w;
    *(v4h*)(sX + r * PITCH_X + c4) = hv;
  }
  for (int t = tid; t < 15 * 24; t += 128) {   // zero pad rows 49..63
    int r = TOK + t / 24, c4 = (t - (t / 24) * 24) * 4;
    v4h z = {};
    *(v4h*)(sX + r * PITCH_X + c4) = z;
  }
  // ---- stage w_qkv^T (16B vector copies) ----
  for (int t = tid; t < SZ_W / 16; t += 128)
    ((uint4*)sW)[t] = ((const uint4*)wqkvT)[t];
  // ---- stage pos_emb (13x13) ----
  for (int t = tid; t < 169; t += 128) sPos[t] = pos_emb[t];
  __syncthreads();

  // ---- precompute bias + shift masks (block-uniform 49x49) ----
  for (int t = tid; t < TOK * TOK; t += 128) {
    int i = t / TOK, j = t - i * TOK;
    int ri = i / 7, ci = i - ri * 7;
    int rj = j / 7, cj = j - rj * 7;
    float v = sPos[(rj - ri + 6) * 13 + (cj - ci + 6)];
    if (maskUL && ((i >= 28) != (j >= 28))) v = -1e30f;
    if (maskLR && ((ci >= 4) != (cj >= 4))) v = -1e30f;
    sBias[i * PITCH_B + j] = v;
  }

  // ---- QKV projection: [64x96] x [96x288]; software-pipelined B tiles ----
  {
    v16h a0 = load_A(sX, PITCH_X, 16 * wave, 0,  lane);
    v16h a1 = load_A(sX, PITCH_X, 16 * wave, 32, lane);
    v16h a2 = load_A(sX, PITCH_X, 16 * wave, 64, lane);
    v16h b0 = load_Brow(sW, DIM, 0, 0,  lane);
    v16h b1 = load_Brow(sW, DIM, 0, 32, lane);
    v16h b2 = load_Brow(sW, DIM, 0, 64, lane);
#pragma unroll
    for (int nt = 0; nt < 18; ++nt) {
      int ntn = (nt + 1) % 18;           // wrap: last iter reloads tile 0 (harmless)
      v16h n0 = load_Brow(sW, DIM, ntn * 16, 0,  lane);
      v16h n1 = load_Brow(sW, DIM, ntn * 16, 32, lane);
      v16h n2 = load_Brow(sW, DIM, ntn * 16, 64, lane);
      v8f acc = {};
      acc = wmma_f16(a0, b0, acc);
      acc = wmma_f16(a1, b1, acc);
      acc = wmma_f16(a2, b2, acc);
      if (nt < 6) {                 // Q: scale by 1/sqrt(d)
        store_D_h(sQKV, PITCH_QKV, 16 * wave, nt * 16, lane, acc, SCALEF);
      } else if (nt < 12) {         // K
        store_D_h(sQKV, PITCH_QKV, 16 * wave, nt * 16, lane, acc, 1.0f);
      } else {                      // V -> transposed [channel][token]
        store_D_hT(sVt, PITCH_VT, 16 * wave, (nt - 12) * 16, lane, acc);
      }
      b0 = n0; b1 = n1; b2 = n2;
    }
  }
  __syncthreads();

  // ---- per-head attention ----
  for (int h = 0; h < HEADS; ++h) {
    // Q*K^T (K^T fragment = contiguous row read of K), pipelined
    {
      v16h aq = load_A(sQKV, PITCH_QKV, 16 * wave, h * HDIM, lane);
      v16h bk = load_Brow(sQKV, PITCH_QKV, 0, DIM + h * HDIM, lane);
#pragma unroll
      for (int nt = 0; nt < 4; ++nt) {
        v16h nb = load_Brow(sQKV, PITCH_QKV, ((nt + 1) & 3) * 16, DIM + h * HDIM, lane);
        v8f acc = {};
        acc = wmma_f16(aq, bk, acc);
        store_D_f(sDots, PITCH_D, 16 * wave, nt * 16, lane, acc);
        bk = nb;
      }
    }
    __syncthreads();
    // softmax rows: unnormalized exp to f16, 1/sum deferred to AV epilogue
    if (tid < 64) {
      int i = tid;
      if (i < TOK) {
        float vmax = -1e30f;
        for (int j = 0; j < TOK; ++j)
          vmax = fmaxf(vmax, sDots[i * PITCH_D + j] + sBias[i * PITCH_B + j]);
        float sum = 0.f;
        for (int j = 0; j < TOK; ++j) {
          float e = __expf(sDots[i * PITCH_D + j] + sBias[i * PITCH_B + j] - vmax);
          sum += e;
          sAttn[i * PITCH_A + j] = (_Float16)e;
        }
        for (int j = TOK; j < 64; ++j) sAttn[i * PITCH_A + j] = (_Float16)0.f;
        sInv[i] = 1.0f / sum;
      } else {
        for (int j = 0; j < 64; ++j) sAttn[i * PITCH_A + j] = (_Float16)0.f;
        sInv[i] = 0.f;
      }
    }
    __syncthreads();
    // attn * V (V^T fragment = contiguous row read), normalize in epilogue
    {
      v16h aa0 = load_A(sAttn, PITCH_A, 16 * wave, 0,  lane);
      v16h aa1 = load_A(sAttn, PITCH_A, 16 * wave, 32, lane);
      v16h bv0 = load_Brow(sVt, PITCH_VT, h * HDIM, 0,  lane);
      v16h bv1 = load_Brow(sVt, PITCH_VT, h * HDIM, 32, lane);
      int m0 = 16 * wave + ((lane >> 4) << 3);
#pragma unroll
      for (int nt = 0; nt < 2; ++nt) {
        v16h nv0 = load_Brow(sVt, PITCH_VT, h * HDIM + ((nt + 1) & 1) * 16, 0,  lane);
        v16h nv1 = load_Brow(sVt, PITCH_VT, h * HDIM + ((nt + 1) & 1) * 16, 32, lane);
        v8f acc = {};
        acc = wmma_f16(aa0, bv0, acc);
        acc = wmma_f16(aa1, bv1, acc);
#pragma unroll
        for (int i = 0; i < 8; ++i) acc[i] *= sInv[m0 + i];
        store_D_h(sCtx, PITCH_C, 16 * wave, h * HDIM + nt * 16, lane, acc, 1.0f);
        bv0 = nv0; bv1 = nv1;
      }
    }
    __syncthreads();
  }

  // ---- stage w_out^T into sW ----
  for (int t = tid; t < (DIM * DIM * 2) / 16; t += 128)
    ((uint4*)sW)[t] = ((const uint4*)woutT)[t];
  __syncthreads();

  // ---- output projection [64x96] x [96x96] + bias, scatter with roll-back ----
  {
    v16h c0 = load_A(sCtx, PITCH_C, 16 * wave, 0,  lane);
    v16h c1 = load_A(sCtx, PITCH_C, 16 * wave, 32, lane);
    v16h c2 = load_A(sCtx, PITCH_C, 16 * wave, 64, lane);
    v16h b0 = load_Brow(sW, DIM, 0, 0,  lane);
    v16h b1 = load_Brow(sW, DIM, 0, 32, lane);
    v16h b2 = load_Brow(sW, DIM, 0, 64, lane);
#pragma unroll
    for (int nt = 0; nt < 6; ++nt) {
      int ntn = (nt + 1) % 6;
      v16h n0 = load_Brow(sW, DIM, ntn * 16, 0,  lane);
      v16h n1 = load_Brow(sW, DIM, ntn * 16, 32, lane);
      v16h n2 = load_Brow(sW, DIM, ntn * 16, 64, lane);
      v8f acc = {};
      acc = wmma_f16(c0, b0, acc);
      acc = wmma_f16(c1, b1, acc);
      acc = wmma_f16(c2, b2, acc);
      int col = nt * 16 + (lane & 15);
      float bias = b_out[col];
      int m0 = 16 * wave + ((lane >> 4) << 3);
#pragma unroll
      for (int i = 0; i < 8; ++i) {
        int m = m0 + i;
        if (m < TOK) {
          int rr = m / 7, cc = m - rr * 7;
          int gh = wy * 7 + rr + 3; if (gh >= IMG) gh -= IMG;
          int gw = wx * 7 + cc + 3; if (gw >= IMG) gw -= IMG;
          out[(((size_t)bb * IMG + gh) * IMG + gw) * DIM + col] = acc[i] + bias;
        }
      }
      b0 = n0; b1 = n1; b2 = n2;
    }
  }
}

// ---------------- channel-attention reductions ----------------
__device__ inline unsigned fenc(float f) {
  unsigned u = __float_as_uint(f);
  return (u & 0x80000000u) ? ~u : (u | 0x80000000u);
}
__device__ inline float fdec(unsigned u) {
  unsigned b = (u & 0x80000000u) ? (u & 0x7fffffffu) : ~u;
  return __uint_as_float(b);
}

__global__ void reduce_kernel(const float* __restrict__ out, float* __restrict__ sums,
                              unsigned* __restrict__ maxe) {
  int bb = blockIdx.y;
  int h0 = blockIdx.x * 8;
  int c = threadIdx.x;  // 0..95
  float s = 0.f, mx = -1e30f;
  for (int hh = h0; hh < h0 + 8; ++hh)
    for (int w = 0; w < IMG; ++w) {
      float v = out[(((size_t)bb * IMG + hh) * IMG + w) * DIM + c];
      s += v;
      mx = fmaxf(mx, v);
    }
  atomicAdd(&sums[bb * DIM + c], s);
  atomicMax(&maxe[bb * DIM + c], fenc(mx));
}

__global__ void gate_kernel(const float* __restrict__ sums, const unsigned* __restrict__ maxe,
                            const float* __restrict__ w11, const float* __restrict__ b11,
                            const float* __restrict__ w12, const float* __restrict__ b12,
                            const float* __restrict__ w21, const float* __restrict__ b21,
                            const float* __restrict__ w22, const float* __restrict__ b22,
                            float* __restrict__ gate) {
  __shared__ float avg[DIM], mx[DIM], h1[DIM], h2[DIM];
  int bb = blockIdx.x;
  int t = threadIdx.x;  // 0..95
  avg[t] = sums[bb * DIM + t] * (1.0f / (IMG * IMG));
  mx[t]  = fdec(maxe[bb * DIM + t]);
  __syncthreads();
  float s1 = b11[t], s2 = b21[t];
  for (int c = 0; c < DIM; ++c) {
    s1 += avg[c] * w11[c * DIM + t];
    s2 += mx[c]  * w21[c * DIM + t];
  }
  h1[t] = fmaxf(s1, 0.f);
  h2[t] = fmaxf(s2, 0.f);
  __syncthreads();
  float a = b12[t], m = b22[t];
  for (int j = 0; j < DIM; ++j) {
    a += h1[j] * w12[j * DIM + t];
    m += h2[j] * w22[j * DIM + t];
  }
  gate[bb * DIM + t] = 1.0f / (1.0f + __expf(-(a + m)));
}

__global__ void apply_gate_kernel(float* __restrict__ out, const float* __restrict__ gate, int n) {
  int idx = blockIdx.x * blockDim.x + threadIdx.x;
  if (idx < n) {
    int c = idx % DIM;
    int bb = idx / (IMG * IMG * DIM);
    out[idx] *= gate[bb * DIM + c];
  }
}

extern "C" void kernel_launch(void* const* d_in, const int* in_sizes, int n_in,
                              void* d_out, int out_size, void* d_ws, size_t ws_size,
                              hipStream_t stream) {
  (void)in_sizes; (void)n_in; (void)out_size; (void)ws_size;
  const float* x      = (const float*)d_in[0];
  const float* w_qkv  = (const float*)d_in[1];
  const float* pos    = (const float*)d_in[2];
  const float* w_out  = (const float*)d_in[3];
  const float* b_out  = (const float*)d_in[4];
  const float* m1w1   = (const float*)d_in[5];
  const float* m1b1   = (const float*)d_in[6];
  const float* m1w2   = (const float*)d_in[7];
  const float* m1b2   = (const float*)d_in[8];
  const float* m2w1   = (const float*)d_in[9];
  const float* m2b1   = (const float*)d_in[10];
  const float* m2w2   = (const float*)d_in[11];
  const float* m2b2   = (const float*)d_in[12];
  float* out = (float*)d_out;

  char* ws = (char*)d_ws;
  _Float16* wqkvT = (_Float16*)ws;                   // 55296 B
  _Float16* woutT = (_Float16*)(ws + 55296);         // 18432 B
  float*    sums  = (float*)   (ws + 73728);         // 3072 B
  unsigned* maxe  = (unsigned*)(ws + 76800);         // 3072 B
  float*    gate  = (float*)   (ws + 79872);         // 3072 B

  prep_kernel<<<108, 256, 0, stream>>>(w_qkv, w_out, wqkvT, woutT, sums, maxe);

  swin_attn_kernel<<<8 * 1024, 128, SMEM_TOTAL, stream>>>(x, pos, b_out, wqkvT,
                                                          woutT, out);

  reduce_kernel<<<dim3(28, 8), 96, 0, stream>>>(out, sums, maxe);

  gate_kernel<<<8, 96, 0, stream>>>(sums, maxe, m1w1, m1b1, m1w2, m1b2,
                                    m2w1, m2b1, m2w2, m2b2, gate);

  const int n = 8 * IMG * IMG * DIM;
  apply_gate_kernel<<<(n + 255) / 256, 256, 0, stream>>>(out, gate, n);
}
